// HieraGLightDQN_28819230556631
// MI455X (gfx1250) — matrix operand order repaired and checked by reading, not compiled, not run on hardware
//
#include <hip/hip_runtime.h>
#include <hip/hip_bf16.h>

typedef __attribute__((ext_vector_type(16))) _Float16 v16h;
typedef __attribute__((ext_vector_type(8)))  _Float16 v8h;
typedef __attribute__((ext_vector_type(4)))  _Float16 v4h;
typedef __attribute__((ext_vector_type(8)))  float    v8f;

#define HDIM 64
#define MDIM 16
#define LDS_STRIDE 72   // 64 + 8 halfs pad to spread LDS banks (144B row, 16B aligned)

__device__ __forceinline__ v8f wmma_f16(v16h a, v16h b, v8f c) {
  // D = A(16x32 f16) * B(32x16 f16) + C(16x16 f32)
  return __builtin_amdgcn_wmma_f32_16x16x32_f16(false, a, false, b, (short)0, c,
                                                false, false);
}

// ---------------------------------------------------------------- zero fill
__global__ void hgl_zero_f32(float* __restrict__ p, long n) {
  long i = (long)blockIdx.x * blockDim.x + threadIdx.x;
  if (i < n) p[i] = 0.0f;
}

// ------------------------------------------------- stage 1: movement MLP
// h = relu(relu(X @ W1 + b1) @ W2 + b2), X:[NM,16], out f16 [NM,64]
__global__ void hgl_movement_mlp(const float* __restrict__ x,
                                 const float* __restrict__ W1, const float* __restrict__ b1,
                                 const float* __restrict__ W2, const float* __restrict__ b2,
                                 _Float16* __restrict__ hout, long NM, int ntiles) {
  __shared__ __align__(16) _Float16 lds[4 * 16 * LDS_STRIDE];
  const int lane       = threadIdx.x & 31;
  const int waveInBlk  = threadIdx.x >> 5;
  const int wavesPerBl = blockDim.x >> 5;
  const int gwave      = blockIdx.x * wavesPerBl + waveInBlk;
  const int nwaves     = gridDim.x * wavesPerBl;
  const int m = lane & 15, half = lane >> 4;
  _Float16* myl = lds + waveInBlk * 16 * LDS_STRIDE;

  // --- hoist weight fragments (B: col N = lane%16, element j holds table-order K) ---
  v16h B1[4];
  #pragma unroll
  for (int nb = 0; nb < 4; ++nb) {
    v16h b = {};
    {                                        // K rows 16..31 are zero pad
      const float* wc = W1 + nb * 16 + m;
      #pragma unroll
      for (int j = 0; j < 8; ++j) b[j] = (_Float16)wc[(half * 8 + j) * HDIM];
    }
    B1[nb] = b;
  }
  v16h B2[2][4];
  #pragma unroll
  for (int kc = 0; kc < 2; ++kc)
    #pragma unroll
    for (int nb = 0; nb < 4; ++nb) {
      const float* wc = W2 + (kc * 32) * HDIM + nb * 16 + m;
      v16h b;
      #pragma unroll
      for (int j = 0; j < 16; ++j) b[j] = (_Float16)wc[((j < 8 ? 0 : 16) + half * 8 + (j & 7)) * HDIM];
      B2[kc][nb] = b;
    }
  float b1s[4], b2s[4];
  #pragma unroll
  for (int nb = 0; nb < 4; ++nb) {
    b1s[nb] = b1[nb * 16 + m];
    b2s[nb] = b2[nb * 16 + m];
  }

  for (int tile = gwave; tile < ntiles; tile += nwaves) {
    long row0 = (long)tile * 16;
    long rA = row0 + m; if (rA >= NM) rA = NM - 1;
    // A layer1: lane holds A[m][half*8 + 0..7], K=16..31 zero
    const float* xr = x + rA * MDIM + half * 8;
    v16h a1 = {};
    #pragma unroll
    for (int j = 0; j < 8; ++j) a1[j] = (_Float16)xr[j];

    // layer 1 -> relu -> LDS (f16, row-major 16x64)
    #pragma unroll
    for (int nb = 0; nb < 4; ++nb) {
      v8f c = {};
      c = wmma_f16(a1, B1[nb], c);
      #pragma unroll
      for (int v = 0; v < 8; ++v) {
        float r = fmaxf(c[v] + b1s[nb], 0.0f);
        myl[(v + 8 * half) * LDS_STRIDE + nb * 16 + m] = (_Float16)r;
      }
    }
    asm volatile("s_wait_dscnt 0" ::: "memory");

    // layer 2 A fragments from LDS (K=64 -> two K=32 chunks)
    v16h a2[2];
    #pragma unroll
    for (int kc = 0; kc < 2; ++kc) {
      const _Float16* lr = myl + m * LDS_STRIDE + kc * 32;
      v8h ra = *(const v8h*)(lr + half * 8);
      v8h rb = *(const v8h*)(lr + 16 + half * 8);
      v16h a;
      #pragma unroll
      for (int j = 0; j < 8; ++j) { a[j] = ra[j]; a[8 + j] = rb[j]; }
      a2[kc] = a;
    }
    asm volatile("s_wait_dscnt 0" ::: "memory");  // a2 loads landed before LDS reuse

    if (row0 + 16 <= NM) {
      // fast path: layer 2 -> relu -> LDS, then 128-bit vector stores to global
      #pragma unroll
      for (int nb = 0; nb < 4; ++nb) {
        v8f c = {};
        c = wmma_f16(a2[0], B2[0][nb], c);
        c = wmma_f16(a2[1], B2[1][nb], c);
        #pragma unroll
        for (int v = 0; v < 8; ++v) {
          float r = fmaxf(c[v] + b2s[nb], 0.0f);
          myl[(v + 8 * half) * LDS_STRIDE + nb * 16 + m] = (_Float16)r;
        }
      }
      asm volatile("s_wait_dscnt 0" ::: "memory");
      // lane copies a contiguous 64B slice: row m, halfs [half*32 .. half*32+31]
      const _Float16* lr = myl + m * LDS_STRIDE + half * 32;
      _Float16* gout = hout + (row0 + m) * HDIM + half * 32;
      #pragma unroll
      for (int q = 0; q < 4; ++q)
        *(v8h*)(gout + q * 8) = *(const v8h*)(lr + q * 8);
      asm volatile("s_wait_dscnt 0" ::: "memory");
    } else {
      // tail tile: guarded scalar stores
      #pragma unroll
      for (int nb = 0; nb < 4; ++nb) {
        v8f c = {};
        c = wmma_f16(a2[0], B2[0][nb], c);
        c = wmma_f16(a2[1], B2[1][nb], c);
        #pragma unroll
        for (int v = 0; v < 8; ++v) {
          long row = row0 + v + 8 * half;
          if (row < NM) {
            float r = fmaxf(c[v] + b2s[nb], 0.0f);
            hout[row * HDIM + nb * 16 + m] = (_Float16)r;
          }
        }
      }
    }
  }
}

// ------------------------------------------------- stage 2: segment max
// pd[dst] = max over edges of h[src]; h >= 0 so uint-bit atomicMax == float max,
// and zero-init reproduces the reference's empty-segment -> 0 behavior.
__global__ void hgl_segmax(const _Float16* __restrict__ h,
                           const int* __restrict__ esrc, const int* __restrict__ edst,
                           unsigned* __restrict__ pd_bits, long EMP) {
  long t = (long)blockIdx.x * blockDim.x + threadIdx.x;
  long e = t >> 4;
  if (e >= EMP) return;
  int c4 = (int)(t & 15) * 4;
  long src = esrc[e], dst = edst[e];
  v4h vals = *(const v4h*)(h + src * HDIM + c4);
  unsigned* out = pd_bits + dst * HDIM + c4;
  #pragma unroll
  for (int i = 0; i < 4; ++i) {
    float f = (float)vals[i];
    atomicMax(&out[i], __float_as_uint(f));
  }
}

// pd (f32) -> f16 for the competition-stage gathers
__global__ void hgl_cvt_pd(const float* __restrict__ pd, _Float16* __restrict__ pdh, long n) {
  long i = (long)blockIdx.x * blockDim.x + threadIdx.x;
  if (i < n) pdh[i] = (_Float16)pd[i];
}

// ------------------------------------------------- stage 3: phase competition
// per edge: m = relu(concat(pd[dst], pd[src]) @ Wc1 + bc1); s[dst] += m; cnt[dst]++
__global__ void hgl_competition(const _Float16* __restrict__ pdh,
                                const int* __restrict__ esrc, const int* __restrict__ edst,
                                const float* __restrict__ Wc1, const float* __restrict__ bc1,
                                float* __restrict__ sbuf, float* __restrict__ cnt,
                                int EPP, int ntiles) {
  const int lane       = threadIdx.x & 31;
  const int waveInBlk  = threadIdx.x >> 5;
  const int wavesPerBl = blockDim.x >> 5;
  const int gwave      = blockIdx.x * wavesPerBl + waveInBlk;
  const int nwaves     = gridDim.x * wavesPerBl;
  const int m = lane & 15, half = lane >> 4;

  // hoist Wc1 fragments: K=128 -> 4 chunks, N=64 -> 4 blocks
  v16h BC[4][4];
  #pragma unroll
  for (int kc = 0; kc < 4; ++kc)
    #pragma unroll
    for (int nb = 0; nb < 4; ++nb) {
      const float* wc = Wc1 + (kc * 32) * HDIM + nb * 16 + m;
      v16h b;
      #pragma unroll
      for (int j = 0; j < 16; ++j) b[j] = (_Float16)wc[((j < 8 ? 0 : 16) + half * 8 + (j & 7)) * HDIM];
      BC[kc][nb] = b;
    }
  float bcs[4];
  #pragma unroll
  for (int nb = 0; nb < 4; ++nb) bcs[nb] = bc1[nb * 16 + m];

  for (int tile = gwave; tile < ntiles; tile += nwaves) {
    int e = tile * 16 + m; if (e >= EPP) e = EPP - 1;
    const _Float16* drow = pdh + (long)edst[e] * HDIM;
    const _Float16* srow = pdh + (long)esrc[e] * HDIM;
    // A: pair[m][k], k<64 -> dst row, k>=64 -> src row; contiguous 16B runs
    v16h A[4];
    #pragma unroll
    for (int kc = 0; kc < 4; ++kc) {
      int c0 = kc * 32;
      const _Float16* base = (c0 < 64) ? (drow + c0) : (srow + (c0 - 64));
      v8h ra = *(const v8h*)(base + half * 8);
      v8h rb = *(const v8h*)(base + 16 + half * 8);
      v16h a;
      #pragma unroll
      for (int j = 0; j < 8; ++j) { a[j] = ra[j]; a[8 + j] = rb[j]; }
      A[kc] = a;
    }
    #pragma unroll
    for (int nb = 0; nb < 4; ++nb) {
      v8f c = {};
      c = wmma_f16(A[0], BC[0][nb], c);
      c = wmma_f16(A[1], BC[1][nb], c);
      c = wmma_f16(A[2], BC[2][nb], c);
      c = wmma_f16(A[3], BC[3][nb], c);
      int col = nb * 16 + m;
      #pragma unroll
      for (int v = 0; v < 8; ++v) {
        int er = tile * 16 + v + 8 * half;
        if (er < EPP) {
          float r = fmaxf(c[v] + bcs[nb], 0.0f);
          atomicAdd(&sbuf[(long)edst[er] * HDIM + col], r);
        }
      }
    }
    int e2 = tile * 16 + lane;
    if (lane < 16 && e2 < EPP) atomicAdd(&cnt[edst[e2]], 1.0f);
  }
}

// ------------------------------------------------- stage 4: q-values
__global__ void hgl_qfinal(const float* __restrict__ sbuf, const float* __restrict__ cnt,
                           const float* __restrict__ Wq, const float* __restrict__ bq,
                           float* __restrict__ q, long NP) {
  long gw = ((long)blockIdx.x * blockDim.x + threadIdx.x) >> 5;
  int lane = threadIdx.x & 31;
  if (gw >= NP) return;
  const float* row = sbuf + gw * HDIM;
  float v = row[lane] * Wq[lane] + row[32 + lane] * Wq[32 + lane];
  #pragma unroll
  for (int off = 16; off > 0; off >>= 1) v += __shfl_xor(v, off, 32);
  if (lane == 0) q[gw] = v / fmaxf(cnt[gw], 1.0f) + bq[0];
}

extern "C" void kernel_launch(void* const* d_in, const int* in_sizes, int n_in,
                              void* d_out, int out_size, void* d_ws, size_t ws_size,
                              hipStream_t stream) {
  const float* x    = (const float*)d_in[0];
  const int*  emp_s = (const int*)d_in[1];
  const int*  emp_d = (const int*)d_in[2];
  const int*  epp_s = (const int*)d_in[3];
  const int*  epp_d = (const int*)d_in[4];
  const float* W1   = (const float*)d_in[5];
  const float* b1   = (const float*)d_in[6];
  const float* W2   = (const float*)d_in[7];
  const float* b2   = (const float*)d_in[8];
  const float* Wc1  = (const float*)d_in[9];
  const float* bc1  = (const float*)d_in[10];
  const float* Wq   = (const float*)d_in[11];
  const float* bq   = (const float*)d_in[12];

  const long NM  = (long)in_sizes[0] / MDIM;
  const long EMP = (long)in_sizes[1];
  const long EPP = (long)in_sizes[3];
  const long NP  = (long)out_size;

  // workspace layout
  char* ws = (char*)d_ws;
  size_t off = 0;
  auto take = [&](size_t bytes) -> char* {
    char* p = ws + off;
    off += (bytes + 255) & ~(size_t)255;
    return p;
  };
  _Float16* h16  = (_Float16*)take(NM * HDIM * sizeof(_Float16));
  float*    pd   = (float*)   take(NP * HDIM * sizeof(float));
  _Float16* pdh  = (_Float16*)take(NP * HDIM * sizeof(_Float16));
  float*    sbuf = (float*)   take(NP * HDIM * sizeof(float));
  float*    cnt  = (float*)   take(NP * sizeof(float));
  (void)ws_size;

  // zero pd, s, cnt (pd zero == float 0 bit pattern; valid max identity for ReLU data)
  {
    long nz = NP * HDIM;
    hgl_zero_f32<<<(unsigned)((nz + 255) / 256), 256, 0, stream>>>(pd, nz);
    hgl_zero_f32<<<(unsigned)((nz + 255) / 256), 256, 0, stream>>>(sbuf, nz);
    hgl_zero_f32<<<(unsigned)((NP + 255) / 256), 256, 0, stream>>>(cnt, NP);
  }

  // stage 1: movement MLP (WMMA)
  {
    int ntiles = (int)((NM + 15) / 16);
    int blocks = (ntiles + 3) / 4; if (blocks > 1024) blocks = 1024;
    hgl_movement_mlp<<<blocks, 128, 0, stream>>>(x, W1, b1, W2, b2, h16, NM, ntiles);
  }

  // stage 2: segment max + f16 conversion
  {
    long t = EMP * 16;
    hgl_segmax<<<(unsigned)((t + 255) / 256), 256, 0, stream>>>(
        h16, emp_s, emp_d, (unsigned*)pd, EMP);
    long n = NP * HDIM;
    hgl_cvt_pd<<<(unsigned)((n + 255) / 256), 256, 0, stream>>>(pd, pdh, n);
  }

  // stage 3: phase competition (WMMA) + segment sum via atomics
  {
    int ntiles = (int)((EPP + 15) / 16);
    int blocks = (ntiles + 3) / 4; if (blocks > 1024) blocks = 1024;
    hgl_competition<<<blocks, 128, 0, stream>>>(pdh, epp_s, epp_d, Wc1, bc1,
                                                sbuf, cnt, (int)EPP, ntiles);
  }

  // stage 4: mean + final linear
  {
    long t = NP * 32;
    hgl_qfinal<<<(unsigned)((t + 255) / 256), 256, 0, stream>>>(
        sbuf, cnt, Wq, bq, (float*)d_out, NP);
  }
}
